// DeepSeekV3Module_74036646248850
// MI455X (gfx1250) — matrix operand order repaired and checked by reading, not compiled, not run on hardware
//
#include <hip/hip_runtime.h>
#include <hip/hip_bf16.h>
#include <math.h>

// ---------------------------------------------------------------------------
// Types for CDNA5 WMMA (gfx1250, wave32)
// ---------------------------------------------------------------------------
typedef __attribute__((ext_vector_type(16))) __bf16 v16bf;
typedef __attribute__((ext_vector_type(8)))  __bf16 bf16x8;
typedef __attribute__((ext_vector_type(8)))  float  v8f;
typedef __attribute__((ext_vector_type(4)))  unsigned u32x4;
typedef __attribute__((ext_vector_type(8)))  unsigned u32x8;

#define D_    1024
#define LAT_  512
#define HD_   64
#define KPH_  32
#define H_    16
#define E_    8
#define SEQ_  2048
#define BT_   4096   // B*S tokens

static __device__ __forceinline__ v8f vzero8() {
  v8f v;
#pragma unroll
  for (int i = 0; i < 8; ++i) v[i] = 0.0f;
  return v;
}

static __device__ __forceinline__ v16bf frag16(const __bf16* lo, const __bf16* hi) {
  bf16x8 a = *(const bf16x8*)lo;
  bf16x8 b = *(const bf16x8*)hi;
  return __builtin_shufflevector(a, b, 0,1,2,3,4,5,6,7,8,9,10,11,12,13,14,15);
}

static __device__ __forceinline__ v8f wmma_bf16(v16bf a, v16bf b, v8f c) {
  return __builtin_amdgcn_wmma_f32_16x16x32_bf16(false, a, false, b, (short)0, c,
                                                 false, false);
}

// ---------------------------------------------------------------------------
// CDNA5 async global->LDS copy (ASYNCcnt-tracked, bypasses VGPRs)
// ---------------------------------------------------------------------------
static __device__ __forceinline__ unsigned ldsoff(const void* p) {
  return (unsigned)(size_t)p;  // low 32 bits of flat LDS-aperture addr = LDS offset
}
static __device__ __forceinline__ void async_ld_b128(unsigned lds, const void* g) {
  asm volatile("global_load_async_to_lds_b128 %0, %1, off"
               :: "v"(lds), "v"((unsigned long long)(size_t)g)
               : "memory");
}
static __device__ __forceinline__ void wait_async() {
  asm volatile("s_wait_asynccnt 0x0" ::: "memory");
}

// ---------------------------------------------------------------------------
// Tensor Data Mover: DMA one 128x32-bf16 tile (row stride = stride_elems) into
// LDS with hardware padding (16 DWORDs + 4 DWORD pad -> 40-element pitch).
// D# layout per CDNA5 ISA. Issued once per wave.
// ---------------------------------------------------------------------------
static __device__ __forceinline__ void tdm_load_tile_128x32(
    unsigned lds, const void* g, unsigned stride_elems) {
  unsigned long long ga = (unsigned long long)(size_t)g;
  u32x4 g0;
  g0[0] = 1u;                                                  // count=1 (valid)
  g0[1] = lds;                                                 // lds_addr (bytes)
  g0[2] = (unsigned)ga;                                        // global_addr[31:0]
  g0[3] = (unsigned)((ga >> 32) & 0x01FFFFFFu) | (2u << 30);   // addr[56:32]|type=2
  u32x8 g1;
  g1[0] = (1u << 16)        // data_size = 1 -> 2 bytes
        | (1u << 20)        // pad_enable
        | (3u << 22)        // pad_interval: 2^(3+1)=16 DWORDs (one 32-elem row)
        | (3u << 25);       // pad_amount: 3+1 = 4 DWORDs (8 elems) -> pitch 40
  g1[1] = 32u << 16;        // tensor_dim0[15:0]=32 (atomic_barrier_addr=0)
  g1[2] = 128u << 16;       // tensor_dim0[31:16]=0 | tensor_dim1[15:0]=128
  g1[3] = 32u << 16;        // tensor_dim1[31:16]=0 | tile_dim0=32
  g1[4] = 128u;             // tile_dim1=128 | tile_dim2=0
  g1[5] = stride_elems;     // tensor_dim0_stride[31:0]
  g1[6] = 0u;               // stride0[47:32]=0 | stride1[15:0]=0
  g1[7] = 0u;               // stride1[47:16]=0 (unused for 2D)
  asm volatile("tensor_load_to_lds %0, %1" :: "s"(g0), "s"(g1) : "memory");
}

// ---------------------------------------------------------------------------
// 16-lane-row butterfly reductions via v_permlane16_b32.
// ---------------------------------------------------------------------------
#if __has_builtin(__builtin_amdgcn_permlane16)
template <unsigned LO, unsigned HI>
static __device__ __forceinline__ float pl16(float x) {
  unsigned u = __float_as_uint(x);
  u = __builtin_amdgcn_permlane16(u, u, LO, HI, false, false);
  return __uint_as_float(u);
}
static __device__ __forceinline__ float red_max16(float x) {
  x = fmaxf(x, pl16<0x67452301u, 0xEFCDAB89u>(x));
  x = fmaxf(x, pl16<0x54761032u, 0xDCFE98BAu>(x));
  x = fmaxf(x, pl16<0x32107654u, 0xBA98FEDCu>(x));
  x = fmaxf(x, pl16<0xFEDCBA98u, 0x76543210u>(x));
  return x;
}
static __device__ __forceinline__ float red_sum16(float x) {
  x += pl16<0x67452301u, 0xEFCDAB89u>(x);
  x += pl16<0x54761032u, 0xDCFE98BAu>(x);
  x += pl16<0x32107654u, 0xBA98FEDCu>(x);
  x += pl16<0xFEDCBA98u, 0x76543210u>(x);
  return x;
}
#else
static __device__ __forceinline__ float red_max16(float x) {
#pragma unroll
  for (int off = 1; off <= 8; off <<= 1) x = fmaxf(x, __shfl_xor(x, off, 32));
  return x;
}
static __device__ __forceinline__ float red_sum16(float x) {
#pragma unroll
  for (int off = 1; off <= 8; off <<= 1) x += __shfl_xor(x, off, 32);
  return x;
}
#endif

// ---------------------------------------------------------------------------
__global__ void cvt_f32_bf16(const float* __restrict__ in, __bf16* __restrict__ out,
                             int n) {
  int i = blockIdx.x * 256 + threadIdx.x;
  int stride = gridDim.x * 256;
  for (; i < n; i += stride) out[i] = (__bf16)in[i];
}

// Batched tiled transpose + f32->bf16: in [z][K,N] f32 -> out [z][N,K] bf16
__global__ __launch_bounds__(256) void cvt_transpose(const float* __restrict__ in,
                                                     __bf16* __restrict__ out,
                                                     int K, int N) {
  const float* src = in + (size_t)blockIdx.z * K * N;
  __bf16* dst = out + (size_t)blockIdx.z * K * N;
  __shared__ float tile[32][33];
  int n0 = blockIdx.x * 32, k0 = blockIdx.y * 32;
  int tx = threadIdx.x & 31, ty = threadIdx.x >> 5;
#pragma unroll
  for (int i = 0; i < 32; i += 8)
    tile[ty + i][tx] = src[(size_t)(k0 + ty + i) * N + n0 + tx];
  __syncthreads();
#pragma unroll
  for (int i = 0; i < 32; i += 8)
    dst[(size_t)(n0 + ty + i) * K + k0 + tx] = (__bf16)tile[tx][ty + i];
}

// ---------------------------------------------------------------------------
// bf16 WMMA GEMM: C[M,N] = A[M,K] @ Bt[N,K]^T + bias  (B given transposed).
// TDM double-buffered mainloop: wave 0 DMAs tile k+1 while all waves run WMMA
// on tile k. Block tile 128x128, 8 waves (4x2), wave tile 32x64, K-chunk 32.
// MODE 0: f32  MODE 1: bf16  MODE 2: GELU->bf16
// MODE 3: out_f32[m,n] += scale[m*scaleStride]*(acc+bias[n])
// MODE 4: bf16 transposed store out[n*M+m]
// ---------------------------------------------------------------------------
template <int MODE>
__global__ __launch_bounds__(256) void gemm_bf16(
    const __bf16* __restrict__ A, const __bf16* __restrict__ Bt,
    const float* __restrict__ bias, void* __restrict__ out, int M, int N, int K,
    int ldb, const float* __restrict__ scale, int scaleStride) {
  constexpr int LS = 40;  // LDS pitch (elements), produced by TDM padding
  __shared__ __bf16 Al[2][128 * LS];
  __shared__ __bf16 Bl[2][128 * LS];
  const int tid  = threadIdx.x;
  const int lane = tid & 31;
  const int wave = tid >> 5;
  const int l16  = lane & 15;
  const int lh   = lane >> 4;
  const int wm   = (wave >> 1) * 32;
  const int wn   = (wave & 1) * 64;
  const int bM   = blockIdx.y * 128;
  const int bN   = blockIdx.x * 128;

  v8f acc[2][4];
#pragma unroll
  for (int i = 0; i < 2; ++i)
#pragma unroll
    for (int j = 0; j < 4; ++j) acc[i][j] = vzero8();

  const int nk = K >> 5;
  if (wave == 0) {  // prologue: DMA first tiles
    tdm_load_tile_128x32(ldsoff(&Al[0][0]), A + (size_t)bM * K, K);
    tdm_load_tile_128x32(ldsoff(&Bl[0][0]), Bt + (size_t)bN * ldb, ldb);
  }

  for (int kc = 0; kc < nk; ++kc) {
    if (wave == 0) {
      if (kc + 1 < nk) {  // DMA next tiles into the alternate buffer
        int k1 = (kc + 1) << 5;
        tdm_load_tile_128x32(ldsoff(&Al[(kc + 1) & 1][0]),
                             A + (size_t)bM * K + k1, K);
        tdm_load_tile_128x32(ldsoff(&Bl[(kc + 1) & 1][0]),
                             Bt + (size_t)bN * ldb + k1, ldb);
        __builtin_amdgcn_s_wait_tensorcnt(2);  // in-order: current pair done
      } else {
        __builtin_amdgcn_s_wait_tensorcnt(0);
      }
    }
    __syncthreads();  // tile kc visible to all waves

    const __bf16* Ab = &Al[kc & 1][0];
    const __bf16* Bb = &Bl[kc & 1][0];
    v16bf aF[2], bF[4];
#pragma unroll
    for (int mi = 0; mi < 2; ++mi) {
      const __bf16* p = Ab + (wm + mi * 16 + l16) * LS + lh * 8;
      aF[mi] = frag16(p, p + 16);
    }
#pragma unroll
    for (int ni = 0; ni < 4; ++ni) {
      const __bf16* p = Bb + (wn + ni * 16 + l16) * LS + lh * 8;
      bF[ni] = frag16(p, p + 16);
    }
#pragma unroll
    for (int mi = 0; mi < 2; ++mi)
#pragma unroll
      for (int ni = 0; ni < 4; ++ni)
        acc[mi][ni] = wmma_bf16(aF[mi], bF[ni], acc[mi][ni]);
    __syncthreads();  // reads done before this buffer is DMA'd again (kc+2)
  }

#pragma unroll
  for (int mi = 0; mi < 2; ++mi)
#pragma unroll
    for (int ni = 0; ni < 4; ++ni) {
      int n = bN + wn + ni * 16 + l16;
      float bv = bias ? bias[n] : 0.0f;
#pragma unroll
      for (int v = 0; v < 8; ++v) {
        int m = bM + wm + mi * 16 + v + lh * 8;
        float x = acc[mi][ni][v] + bv;
        if constexpr (MODE == 0) {
          ((float*)out)[(size_t)m * N + n] = x;
        } else if constexpr (MODE == 1) {
          ((__bf16*)out)[(size_t)m * N + n] = (__bf16)x;
        } else if constexpr (MODE == 2) {
          x = 0.5f * x * (1.0f + erff(x * 0.70710678118654752f));
          ((__bf16*)out)[(size_t)m * N + n] = (__bf16)x;
        } else if constexpr (MODE == 3) {
          ((float*)out)[(size_t)m * N + n] += scale[(size_t)m * scaleStride] * x;
        } else {  // MODE 4: transposed bf16 store
          ((__bf16*)out)[(size_t)n * M + m] = (__bf16)x;
        }
      }
    }
}

// ---------------------------------------------------------------------------
// Flash attention. Block: 8 waves x 16 query rows. KV tiles of 64, staged with
// async global->LDS copies (V pre-transposed so both tiles are straight copies).
// ---------------------------------------------------------------------------
__global__ __launch_bounds__(256) void mla_attn(const __bf16* __restrict__ Q,
                                                const __bf16* __restrict__ Kc,
                                                const __bf16* __restrict__ Vt,
                                                __bf16* __restrict__ AO) {
  __shared__ __bf16 Kl[64 * 40];     // [t][k]
  __shared__ __bf16 Vl[32 * 72];     // [n][t]
  __shared__ __bf16 Pl[8][16 * 72];  // per-wave P relayout buffer
  const int tid  = threadIdx.x;
  const int lane = tid & 31;
  const int wave = tid >> 5;
  const int l16  = lane & 15;
  const int lh   = lane >> 4;
  const int b  = blockIdx.z;
  const int h  = blockIdx.y;
  const int q0 = blockIdx.x * 128;
  const size_t base = (size_t)b * SEQ_;

  const int qr = q0 + wave * 16 + l16;
  const __bf16* qp = Q + (base + qr) * D_ + h * HD_ + lh * 8;
  v16bf qF = frag16(qp, qp + 16);

  float mrun[8], lrun[8];
  v8f accO[2];
#pragma unroll
  for (int v = 0; v < 8; ++v) { mrun[v] = -1e30f; lrun[v] = 0.0f; }
  accO[0] = vzero8();
  accO[1] = vzero8();
  const float sc = 0.17677669529663687f;  // 1/sqrt(32)

  for (int t0 = 0; t0 < SEQ_; t0 += 64) {
    {
      int idx = tid * 8;
      int r = idx >> 5, c = idx & 31;
      async_ld_b128(ldsoff(Kl + r * 40 + c),
                    Kc + (base + t0 + r) * LAT_ + h * KPH_ + c);
      int r2 = idx >> 6, c2 = idx & 63;
      async_ld_b128(ldsoff(Vl + r2 * 72 + c2),
                    Vt + (size_t)(h * KPH_ + r2) * BT_ + base + t0 + c2);
    }
    wait_async();
    __syncthreads();

    v8f S[4];
#pragma unroll
    for (int ni = 0; ni < 4; ++ni) {
      const __bf16* kp = Kl + (ni * 16 + l16) * 40 + lh * 8;
      v16bf kF = frag16(kp, kp + 16);
      S[ni] = wmma_bf16(qF, kF, vzero8());
    }

    float mt[8];
#pragma unroll
    for (int v = 0; v < 8; ++v)
      mt[v] = red_max16(sc * fmaxf(fmaxf(S[0][v], S[1][v]),
                                   fmaxf(S[2][v], S[3][v])));
    float corr[8], rs[8];
#pragma unroll
    for (int v = 0; v < 8; ++v) {
      float mn = fmaxf(mrun[v], mt[v]);
      corr[v]  = __expf(mrun[v] - mn);
      mrun[v]  = mn;
      rs[v]    = 0.0f;
    }
#pragma unroll
    for (int ni = 0; ni < 4; ++ni)
#pragma unroll
      for (int v = 0; v < 8; ++v) {
        float p = __expf(S[ni][v] * sc - mrun[v]);
        S[ni][v] = p;
        rs[v] += p;
      }
#pragma unroll
    for (int v = 0; v < 8; ++v) lrun[v] = lrun[v] * corr[v] + red_sum16(rs[v]);
#pragma unroll
    for (int nf = 0; nf < 2; ++nf)
#pragma unroll
      for (int v = 0; v < 8; ++v) accO[nf][v] *= corr[v];

    __bf16* pw = &Pl[wave][0];
#pragma unroll
    for (int ni = 0; ni < 4; ++ni)
#pragma unroll
      for (int v = 0; v < 8; ++v)
        pw[(v + lh * 8) * 72 + ni * 16 + l16] = (__bf16)S[ni][v];
    __syncthreads();

#pragma unroll
    for (int kc = 0; kc < 2; ++kc) {
      const __bf16* pp = pw + l16 * 72 + kc * 32 + lh * 8;
      v16bf pF = frag16(pp, pp + 16);
#pragma unroll
      for (int nf = 0; nf < 2; ++nf) {
        const __bf16* vp = Vl + (nf * 16 + l16) * 72 + kc * 32 + lh * 8;
        v16bf vF = frag16(vp, vp + 16);
        accO[nf] = wmma_bf16(pF, vF, accO[nf]);
      }
    }
    __syncthreads();
  }

#pragma unroll
  for (int nf = 0; nf < 2; ++nf) {
    int n = h * KPH_ + nf * 16 + l16;
#pragma unroll
    for (int v = 0; v < 8; ++v) {
      int row = q0 + wave * 16 + v + lh * 8;
      AO[(base + row) * LAT_ + n] = (__bf16)(accO[nf][v] / lrun[v]);
    }
  }
}

// ---------------------------------------------------------------------------
__global__ void router_kernel(const float* __restrict__ mla,
                              const float* __restrict__ rw,
                              const float* __restrict__ rb,
                              float* __restrict__ comb) {
  int t = blockIdx.x * blockDim.x + threadIdx.x;
  if (t >= BT_) return;
  float lg[E_];
#pragma unroll
  for (int e = 0; e < E_; ++e) lg[e] = rb[e];
  const float* x = mla + (size_t)t * D_;
  for (int d = 0; d < D_; ++d) {
    float xv = x[d];
#pragma unroll
    for (int e = 0; e < E_; ++e) lg[e] += xv * rw[d * E_ + e];
  }
  float mx = lg[0];
#pragma unroll
  for (int e = 1; e < E_; ++e) mx = fmaxf(mx, lg[e]);
  float p[E_], s = 0.0f;
#pragma unroll
  for (int e = 0; e < E_; ++e) { p[e] = __expf(lg[e] - mx); s += p[e]; }
#pragma unroll
  for (int e = 0; e < E_; ++e) p[e] /= s;
  int i1 = 0;
#pragma unroll
  for (int e = 1; e < E_; ++e) if (p[e] > p[i1]) i1 = e;
  int i2 = (i1 == 0) ? 1 : 0;
#pragma unroll
  for (int e = 0; e < E_; ++e) if (e != i1 && p[e] > p[i2]) i2 = e;
  float norm = p[i1] + p[i2] + 1e-8f;
#pragma unroll
  for (int e = 0; e < E_; ++e) comb[(size_t)t * E_ + e] = 0.0f;
  comb[(size_t)t * E_ + i1] = p[i1] / norm;
  comb[(size_t)t * E_ + i2] = p[i2] / norm;
}

__global__ void final_add(const float* __restrict__ hs, const float* __restrict__ moe,
                          float* __restrict__ out, int n) {
  int i = blockIdx.x * 256 + threadIdx.x;
  int stride = gridDim.x * 256;
  for (; i < n; i += stride) out[i] = hs[i] + 0.3f * moe[i];
}

// ---------------------------------------------------------------------------
// Host launcher
// ---------------------------------------------------------------------------
extern "C" void kernel_launch(void* const* d_in, const int* in_sizes, int n_in,
                              void* d_out, int out_size, void* d_ws, size_t ws_size,
                              hipStream_t stream) {
  const float* hs       = (const float*)d_in[0];
  const float* q_w      = (const float*)d_in[1];
  const float* q_b      = (const float*)d_in[2];
  const float* k_w      = (const float*)d_in[3];
  const float* k_b      = (const float*)d_in[4];
  const float* v_w      = (const float*)d_in[5];
  const float* v_b      = (const float*)d_in[6];
  const float* o_w      = (const float*)d_in[7];
  const float* o_b      = (const float*)d_in[8];
  const float* router_w = (const float*)d_in[9];
  const float* router_b = (const float*)d_in[10];
  const float* w1       = (const float*)d_in[11];
  const float* b1       = (const float*)d_in[12];
  const float* w2       = (const float*)d_in[13];
  const float* b2       = (const float*)d_in[14];
  float* out = (float*)d_out;

  char* wsp = (char*)d_ws;
  auto alloc = [&](size_t bytes) -> char* {
    char* p = wsp;
    wsp += (bytes + 255) & ~(size_t)255;
    return p;
  };
  __bf16* hs_bf  = (__bf16*)alloc((size_t)BT_ * D_ * 2);
  __bf16* qw_t   = (__bf16*)alloc((size_t)D_ * D_ * 2);
  __bf16* kw_t   = (__bf16*)alloc((size_t)D_ * LAT_ * 2);
  __bf16* vw_t   = (__bf16*)alloc((size_t)D_ * LAT_ * 2);
  __bf16* ow_t   = (__bf16*)alloc((size_t)D_ * D_ * 2);
  __bf16* w1_t   = (__bf16*)alloc((size_t)E_ * D_ * 2 * D_ * 2);
  __bf16* w2_t   = (__bf16*)alloc((size_t)E_ * 2 * D_ * D_ * 2);
  __bf16* Qbf    = (__bf16*)alloc((size_t)BT_ * D_ * 2);
  __bf16* Kbf    = (__bf16*)alloc((size_t)BT_ * LAT_ * 2);
  __bf16* Vt     = (__bf16*)alloc((size_t)LAT_ * BT_ * 2);
  __bf16* aobf   = (__bf16*)alloc((size_t)BT_ * LAT_ * 2);
  float*  mla    = (float*)alloc((size_t)BT_ * D_ * 4);
  __bf16* mlabf  = (__bf16*)alloc((size_t)BT_ * D_ * 2);
  float*  comb   = (float*)alloc((size_t)BT_ * E_ * 4);
  __bf16* h1bf   = (__bf16*)alloc((size_t)BT_ * 2 * D_ * 2);
  float*  moeacc = (float*)alloc((size_t)BT_ * D_ * 4);

  dim3 blk(256);
  cvt_f32_bf16<<<dim3(8192), blk, 0, stream>>>(hs, hs_bf, BT_ * D_);
  cvt_transpose<<<dim3(D_ / 32, D_ / 32, 1), blk, 0, stream>>>(q_w, qw_t, D_, D_);
  cvt_transpose<<<dim3(LAT_ / 32, D_ / 32, 1), blk, 0, stream>>>(k_w, kw_t, D_, LAT_);
  cvt_transpose<<<dim3(LAT_ / 32, D_ / 32, 1), blk, 0, stream>>>(v_w, vw_t, D_, LAT_);
  cvt_transpose<<<dim3(D_ / 32, D_ / 32, 1), blk, 0, stream>>>(o_w, ow_t, D_, D_);
  cvt_transpose<<<dim3(2 * D_ / 32, D_ / 32, E_), blk, 0, stream>>>(w1, w1_t, D_, 2 * D_);
  cvt_transpose<<<dim3(D_ / 32, 2 * D_ / 32, E_), blk, 0, stream>>>(w2, w2_t, 2 * D_, D_);

  gemm_bf16<1><<<dim3(D_ / 128, BT_ / 128), blk, 0, stream>>>(
      hs_bf, qw_t, q_b, (void*)Qbf, BT_, D_, D_, D_, nullptr, 0);
  gemm_bf16<1><<<dim3(LAT_ / 128, BT_ / 128), blk, 0, stream>>>(
      hs_bf, kw_t, k_b, (void*)Kbf, BT_, LAT_, D_, D_, nullptr, 0);
  gemm_bf16<4><<<dim3(LAT_ / 128, BT_ / 128), blk, 0, stream>>>(
      hs_bf, vw_t, v_b, (void*)Vt, BT_, LAT_, D_, D_, nullptr, 0);

  mla_attn<<<dim3(SEQ_ / 128, H_, 2), blk, 0, stream>>>(Qbf, Kbf, Vt, aobf);

  gemm_bf16<0><<<dim3(D_ / 128, BT_ / 128), blk, 0, stream>>>(
      aobf, ow_t, o_b, (void*)mla, BT_, D_, LAT_, D_, nullptr, 0);
  cvt_f32_bf16<<<dim3(8192), blk, 0, stream>>>(mla, mlabf, BT_ * D_);

  router_kernel<<<dim3((BT_ + 255) / 256), blk, 0, stream>>>(mla, router_w,
                                                             router_b, comb);

  hipMemsetAsync(moeacc, 0, (size_t)BT_ * D_ * 4, stream);
  for (int e = 0; e < E_; ++e) {
    const __bf16* w1e = w1_t + (size_t)e * D_ * 2 * D_;
    const __bf16* w2e = w2_t + (size_t)e * 2 * D_ * D_;
    gemm_bf16<2><<<dim3(2 * D_ / 128, BT_ / 128), blk, 0, stream>>>(
        mlabf, w1e, b1 + (size_t)e * 2 * D_, (void*)h1bf, BT_, 2 * D_, D_, D_,
        nullptr, 0);
    gemm_bf16<3><<<dim3(D_ / 128, BT_ / 128), blk, 0, stream>>>(
        h1bf, w2e, b2 + (size_t)e * D_, (void*)moeacc, BT_, D_, 2 * D_, 2 * D_,
        comb + e, E_);
  }

  final_add<<<dim3(8192), blk, 0, stream>>>(hs, moeacc, out, BT_ * D_);
}
